// PCGTConvLayer_9225589752429
// MI455X (gfx1250) — matrix operand order repaired and checked by reading, not compiled, not run on hardware
//
#include <hip/hip_runtime.h>
#include <math.h>

typedef _Float16 h16;
typedef __attribute__((ext_vector_type(16))) _Float16 v16h;
typedef __attribute__((ext_vector_type(8)))  _Float16 v8h;
typedef __attribute__((ext_vector_type(8)))  float    v8f;
typedef __attribute__((ext_vector_type(4))) unsigned int u32x4;
typedef __attribute__((ext_vector_type(8))) int i32x8;
typedef __attribute__((ext_vector_type(4))) int i32x4;

#define NROWS 32768
#define INC   256
#define HH    4
#define DD    128
#define HD    512   // H*D
#define KPART 64
#define PSZ   512

#define WMMA(a,b,c) __builtin_amdgcn_wmma_f32_16x16x32_f16(false,(a),false,(b),(short)0,(c),false,false)

// CDNA5 async global->LDS path (ASYNCcnt-tracked), guarded so compile never breaks.
#if __has_builtin(__builtin_amdgcn_global_load_async_to_lds_b32)
#define USE_ASYNC 1
#else
#define USE_ASYNC 0
#endif

// Builtin prototype (from hipcc diagnostic): (AS1 int* src, AS3 int* dst, imm, imm cpol)
#define GLOBAL_AS1_INT(p) ((__attribute__((address_space(1))) int*)(p))
#define LDS_AS3_INT(p)    ((__attribute__((address_space(3))) int*)(p))

// Tensor Data Mover path, guarded; arity differs across toolchains (bridge doc).
#if __has_builtin(__builtin_amdgcn_tensor_load_to_lds) && __has_builtin(__builtin_amdgcn_s_wait_tensorcnt)
#define USE_TDM 1
#else
#define USE_TDM 0
#endif
#if __has_include(<hip/amd_detail/amd_gfx1250_TDM.h>)
#define TDM_6ARG 1
#else
#define TDM_6ARG 0
#endif

__device__ __forceinline__ void wait_async_zero() {
#if __has_builtin(__builtin_amdgcn_s_wait_asynccnt)
  __builtin_amdgcn_s_wait_asynccnt(0);
#else
  asm volatile("s_wait_asynccnt 0" ::: "memory");
#endif
}

// ---- fragment helpers ------------------------------------------------------
// A fragment in this file's contiguous k order (B matches): lane m=lane&15 holds K = {16*hs+0..7} in a[0..7]
// and K = {16*hs+8..15} in a[8..15] (hs = lane>>4).
__device__ __forceinline__ v16h load_a_frag(const h16* row_base, int hs) {
  v8h lo = *(const v8h*)(row_base + 16 * hs);
  v8h hi = *(const v8h*)(row_base + 16 * hs + 8);
  v16h a;
#pragma unroll
  for (int i = 0; i < 8; ++i) { a[i] = lo[i]; a[i + 8] = hi[i]; }
  return a;
}

// Row-wise reductions across the 16 lanes holding one C/D row (wave32 halves).
__device__ __forceinline__ float rowmax16(float x) {
  x = fmaxf(x, __shfl_xor(x, 1, 32));
  x = fmaxf(x, __shfl_xor(x, 2, 32));
  x = fmaxf(x, __shfl_xor(x, 4, 32));
  x = fmaxf(x, __shfl_xor(x, 8, 32));
  return x;
}
__device__ __forceinline__ float rowsum16(float x) {
  x += __shfl_xor(x, 1, 32);
  x += __shfl_xor(x, 2, 32);
  x += __shfl_xor(x, 4, 32);
  x += __shfl_xor(x, 8, 32);
  return x;
}

// ---- kernel 0: convert Wq|Wk|Wv (f32 [512,256]) to one f16 buffer ----------
__global__ __launch_bounds__(256) void k_cvtW(const float* __restrict__ wq,
                                              const float* __restrict__ wk,
                                              const float* __restrict__ wv,
                                              h16* __restrict__ W16) {
  int i = blockIdx.x * 256 + threadIdx.x;   // < 3*512*256
  int m = i / (HD * INC);
  int j = i - m * (HD * INC);
  const float* s = (m == 0) ? wq : (m == 1) ? wk : wv;
  W16[i] = (h16)s[j];
}

// ---- kernel 1: fused QKV GEMM  out[n,o] = sum_c x[n,c]*W[o,c] + b[o] -------
__global__ __launch_bounds__(256) void k_qkv(const float* __restrict__ x,
                                             const h16* __restrict__ W16,
                                             const float* __restrict__ bq,
                                             const float* __restrict__ bk,
                                             const float* __restrict__ bv,
                                             h16* __restrict__ Q,
                                             h16* __restrict__ K,
                                             h16* __restrict__ V) {
  __shared__ h16 xs[16][INC] __attribute__((aligned(128)));   // 8 KB
  int mat = blockIdx.x;
  int rb  = blockIdx.y;
  int tid = threadIdx.x, lane = tid & 31, wave = tid >> 5;
  int hs = lane >> 4, nn = lane & 15;

  { // cooperative load+convert of the 16x256 x tile
    int r  = tid >> 4;
    int c0 = (tid & 15) * 16;
    const float* src = x + (size_t)(rb * 16 + r) * INC + c0;
#pragma unroll
    for (int i = 0; i < 16; ++i) xs[r][c0 + i] = (h16)src[i];
  }
  __syncthreads();

  const h16*   Wm   = W16 + (size_t)mat * HD * INC;
  const float* bias = (mat == 0) ? bq : (mat == 1) ? bk : bv;
  h16*         out  = (mat == 0) ? Q : (mat == 1) ? K : V;
  int n0 = wave * 64;

  v8f acc[4];
#pragma unroll
  for (int t = 0; t < 4; ++t)
#pragma unroll
    for (int i = 0; i < 8; ++i) acc[t][i] = 0.f;

#pragma unroll
  for (int ks = 0; ks < 8; ++ks) {
    int k0 = ks * 32;
    v16h a = load_a_frag(&xs[nn][k0], hs);
#pragma unroll
    for (int t = 0; t < 4; ++t) {
      v16h b = *(const v16h*)(Wm + (size_t)(n0 + t * 16 + nn) * INC + k0 + 16 * hs);
      acc[t] = WMMA(a, b, acc[t]);
    }
  }
  int row0 = rb * 16;
#pragma unroll
  for (int t = 0; t < 4; ++t) {
    int col = n0 + t * 16 + nn;
    float bsv = bias[col];
#pragma unroll
    for (int v = 0; v < 8; ++v) {
      int r = row0 + v + 8 * hs;
      out[(size_t)r * HD + col] = (h16)(acc[t][v] + bsv);
    }
  }
}

// ---- kernel 2: fine (partitioned) flash attention --------------------------
// grid 1024 = 64 partitions * 4 heads * 4 query-blocks of 128 rows.
// K chunk gathered straight into LDS with async-to-LDS loads (ASYNCcnt path);
// V staged transposed via the sync path (async cannot transpose).
__global__ __launch_bounds__(256) void k_fine(const h16* __restrict__ Q,
                                              const h16* __restrict__ K,
                                              const h16* __restrict__ V,
                                              const int* __restrict__ pidx,
                                              h16* __restrict__ outf /* [N][H][128] */) {
  __shared__ h16 Kc[32][DD]      __attribute__((aligned(128)));  // 8 KB
  __shared__ h16 Vt[DD][32]      __attribute__((aligned(128)));  // 8 KB (transposed)
  __shared__ h16 Pst[8][16][32]  __attribute__((aligned(128)));  // 8 KB
  int bid  = blockIdx.x;
  int part = bid >> 4;
  int h    = (bid >> 2) & 3;
  int qb   = bid & 3;
  int tid = threadIdx.x, lane = tid & 31, wave = tid >> 5;
  int hs = lane >> 4, nn = lane & 15;
  const float scale = 0.08838834764831845f;  // 1/sqrt(128)

  int qbase  = part * PSZ + qb * 128 + wave * 16;
  int qrow_m = pidx[qbase + nn];
  v16h qf[4];
#pragma unroll
  for (int s = 0; s < 4; ++s)
    qf[s] = load_a_frag(Q + (size_t)qrow_m * HD + h * DD + 32 * s, hs);

  v8f acc[8];
  float mrun[8], lrun[8];
#pragma unroll
  for (int t = 0; t < 8; ++t)
#pragma unroll
    for (int i = 0; i < 8; ++i) acc[t][i] = 0.f;
#pragma unroll
  for (int v = 0; v < 8; ++v) { mrun[v] = -1e30f; lrun[v] = 0.f; }

  for (int ch = 0; ch < 16; ++ch) {
    { // cooperative chunk staging
      int c2 = tid & 63;       // dword column (2 halves)
      int r0 = tid >> 6;       // 0..3
#pragma unroll
      for (int i = 0; i < 8; ++i) {
        int rk   = r0 + 4 * i;
        int grow = pidx[part * PSZ + ch * 32 + rk];
        const h16* kp = K + (size_t)grow * HD + h * DD + 2 * c2;
#if USE_ASYNC
        __builtin_amdgcn_global_load_async_to_lds_b32(
            GLOBAL_AS1_INT(kp), LDS_AS3_INT(&Kc[rk][2 * c2]), 0, 0);
#else
        Kc[rk][2 * c2]     = kp[0];
        Kc[rk][2 * c2 + 1] = kp[1];
#endif
        const h16* vp = V + (size_t)grow * HD + h * DD + 2 * c2;
        Vt[2 * c2][rk]     = vp[0];
        Vt[2 * c2 + 1][rk] = vp[1];
        if (ch < 15) __builtin_prefetch(K + (size_t)pidx[part * PSZ + ch * 32 + 32 + rk] * HD + h * DD, 0, 1);
      }
    }
#if USE_ASYNC
    wait_async_zero();
#endif
    __syncthreads();

    // S strip: 16 queries x 32 keys
    v8f s0, s1;
#pragma unroll
    for (int i = 0; i < 8; ++i) { s0[i] = 0.f; s1[i] = 0.f; }
#pragma unroll
    for (int s = 0; s < 4; ++s) {
      v16h b0 = *(const v16h*)&Kc[nn][32 * s + 16 * hs];
      v16h b1 = *(const v16h*)&Kc[16 + nn][32 * s + 16 * hs];
      s0 = WMMA(qf[s], b0, s0);
      s1 = WMMA(qf[s], b1, s1);
    }
    // streaming softmax update + stage P (f16, A-layout source) in LDS
#pragma unroll
    for (int v = 0; v < 8; ++v) {
      float a0 = s0[v] * scale, a1 = s1[v] * scale;
      float mx = rowmax16(fmaxf(a0, a1));
      float mn = fmaxf(mrun[v], mx);
      float corr = __expf(mrun[v] - mn);
      float p0 = __expf(a0 - mn), p1 = __expf(a1 - mn);
      float rs = rowsum16(p0 + p1);
      lrun[v] = lrun[v] * corr + rs;
      mrun[v] = mn;
#pragma unroll
      for (int t = 0; t < 8; ++t) acc[t][v] *= corr;
      int r = v + 8 * hs;
      Pst[wave][r][nn]      = (h16)p0;
      Pst[wave][r][16 + nn] = (h16)p1;
    }
    // O += P(16x32) @ V(32x128)
    v16h pa = load_a_frag(&Pst[wave][nn][0], hs);
#pragma unroll
    for (int t = 0; t < 8; ++t) {
      v16h b = *(const v16h*)&Vt[t * 16 + nn][16 * hs];
      acc[t] = WMMA(pa, b, acc[t]);
    }
    __syncthreads();
  }
  // epilogue: divide by row sums, store per-head output (no atomics)
#pragma unroll
  for (int v = 0; v < 8; ++v) {
    int grow = pidx[qbase + v + 8 * hs];
    float inv = 1.f / lrun[v];
#pragma unroll
    for (int t = 0; t < 8; ++t)
      outf[((size_t)grow * HH + h) * DD + t * 16 + nn] = (h16)(acc[t][v] * inv);
  }
}

// ---- kernel 3: centroids ck/cv = mean over partition of K/V ----------------
__global__ __launch_bounds__(256) void k_cent(const h16* __restrict__ K,
                                              const h16* __restrict__ V,
                                              const int* __restrict__ pidx,
                                              h16* __restrict__ ck,
                                              h16* __restrict__ cv) {
  int part = blockIdx.x;
  int tid  = threadIdx.x;
  for (int c = tid; c < HD; c += 256) {
    float sk = 0.f, sv = 0.f;
    for (int p = 0; p < PSZ; ++p) {
      int grow = pidx[part * PSZ + p];
      sk += (float)K[(size_t)grow * HD + c];
      sv += (float)V[(size_t)grow * HD + c];
    }
    ck[(size_t)part * HD + c] = (h16)(sk * (1.f / PSZ));
    cv[(size_t)part * HD + c] = (h16)(sv * (1.f / PSZ));
  }
}

// ---- kernel 4: coarse attention over centroids + final combine -------------
// ck head tile (64 x 128 f16, row stride 512) fetched by the Tensor Data Mover
// when available: one wave issues the 2D-tile descriptor, waits TENSORcnt,
// the workgroup barrier publishes the LDS tile.
__global__ __launch_bounds__(256) void k_coarse(const h16* __restrict__ Q,
                                                const h16* __restrict__ V,
                                                const h16* __restrict__ ck,
                                                const h16* __restrict__ cv,
                                                const h16* __restrict__ outf,
                                                const float* __restrict__ alpha_logit,
                                                const float* __restrict__ beta_p,
                                                float* __restrict__ out) {
  __shared__ h16 ckL[64][DD]     __attribute__((aligned(128)));  // 16 KB
  __shared__ h16 cvtL[DD][64]    __attribute__((aligned(128)));  // 16 KB (transposed)
  __shared__ h16 Pst[8][16][64]  __attribute__((aligned(128)));  // 16 KB
  int tid = threadIdx.x, lane = tid & 31, wave = tid >> 5;
  int hs = lane >> 4, nn = lane & 15;
  int n0 = blockIdx.x * 128 + wave * 16;
  const float scale = 0.08838834764831845f;

  v8f xg[8];
#pragma unroll
  for (int t = 0; t < 8; ++t)
#pragma unroll
    for (int i = 0; i < 8; ++i) xg[t][i] = 0.f;

  for (int h = 0; h < HH; ++h) {
    { // cooperative load: cvtL transposed always; ckL via TDM when available
      int c2 = tid & 63, r0 = tid >> 6;
#pragma unroll
      for (int i = 0; i < 16; ++i) {
        int rk = r0 + 4 * i;  // centroid 0..63
#if !USE_TDM
        const h16* kp = ck + (size_t)rk * HD + h * DD + 2 * c2;
        ckL[rk][2 * c2]     = kp[0];
        ckL[rk][2 * c2 + 1] = kp[1];
#endif
        const h16* vp = cv + (size_t)rk * HD + h * DD + 2 * c2;
        cvtL[2 * c2][rk]     = vp[0];
        cvtL[2 * c2 + 1][rk] = vp[1];
      }
#if USE_TDM
      if (tid < 32) {  // wave 0 issues the TDM descriptor (wave-uniform branch)
        unsigned long long ga = (unsigned long long)(const void*)(ck + h * DD);
        unsigned lds_off =
            (unsigned)(size_t)(__attribute__((address_space(3))) h16*)&ckL[0][0];
        u32x4 g0;
        g0[0] = 1u;                                   // count=1, user mode
        g0[1] = lds_off;                              // lds_addr
        g0[2] = (unsigned)(ga & 0xffffffffu);         // global_addr[31:0]
        g0[3] = (unsigned)((ga >> 32) & 0x01ffffffu)  // global_addr[56:32]
                | (2u << 30);                         // type=2 ("image")
        i32x8 g1;
        g1[0] = (int)(1u << 16);     // data_size=1 -> 2-byte elements
        g1[1] = (int)(128u << 16);   // tensor_dim0 = 128 (bits 79:48, low half)
        g1[2] = (int)(64u << 16);    // tensor_dim1 = 64  (bits 111:80, low half)
        g1[3] = (int)(128u << 16);   // tile_dim0 = 128   (bits 127:112)
        g1[4] = 64;                  // tile_dim1 = 64    (bits 143:128)
        g1[5] = 512;                 // tensor_dim0_stride = 512 (bits 207:160)
        g1[6] = 0;
        g1[7] = 0;
        i32x4 z4;
        z4[0] = 0; z4[1] = 0; z4[2] = 0; z4[3] = 0;
#if TDM_6ARG
        i32x8 z8;
#pragma unroll
        for (int q = 0; q < 8; ++q) z8[q] = 0;
        __builtin_amdgcn_tensor_load_to_lds(g0, g1, z4, z4, z8, 0);
#else
        __builtin_amdgcn_tensor_load_to_lds(g0, g1, z4, z4, 0);
#endif
        __builtin_amdgcn_s_wait_tensorcnt(0);
      }
#endif
    }
    __syncthreads();

    v16h qf[4];
#pragma unroll
    for (int s = 0; s < 4; ++s)
      qf[s] = load_a_frag(Q + (size_t)(n0 + nn) * HD + h * DD + 32 * s, hs);

    v8f sc[4];
#pragma unroll
    for (int t = 0; t < 4; ++t)
#pragma unroll
      for (int i = 0; i < 8; ++i) sc[t][i] = 0.f;
#pragma unroll
    for (int nt = 0; nt < 4; ++nt)
#pragma unroll
      for (int s = 0; s < 4; ++s) {
        v16h b = *(const v16h*)&ckL[nt * 16 + nn][32 * s + 16 * hs];
        sc[nt] = WMMA(qf[s], b, sc[nt]);
      }
    // full softmax over 64 centroids; stage normalized P in LDS
#pragma unroll
    for (int v = 0; v < 8; ++v) {
      float mx = -1e30f;
#pragma unroll
      for (int nt = 0; nt < 4; ++nt) mx = fmaxf(mx, sc[nt][v] * scale);
      mx = rowmax16(mx);
      float e[4], sum = 0.f;
#pragma unroll
      for (int nt = 0; nt < 4; ++nt) { e[nt] = __expf(sc[nt][v] * scale - mx); sum += e[nt]; }
      sum = rowsum16(sum);
      float inv = 1.f / sum;
      int r = v + 8 * hs;
#pragma unroll
      for (int nt = 0; nt < 4; ++nt) Pst[wave][r][nt * 16 + nn] = (h16)(e[nt] * inv);
    }
    v16h pa0 = load_a_frag(&Pst[wave][nn][0], hs);
    v16h pa1 = load_a_frag(&Pst[wave][nn][32], hs);
#pragma unroll
    for (int t = 0; t < 8; ++t) {
      v16h b0 = *(const v16h*)&cvtL[t * 16 + nn][16 * hs];
      v16h b1 = *(const v16h*)&cvtL[t * 16 + nn][32 + 16 * hs];
      xg[t] = WMMA(pa0, b0, xg[t]);
      xg[t] = WMMA(pa1, b1, xg[t]);
    }
    __syncthreads();
  }

  float alpha = 1.f / (1.f + __expf(-alpha_logit[0]));
  float beta  = beta_p[0];
#pragma unroll
  for (int v = 0; v < 8; ++v) {
    int r = n0 + v + 8 * hs;
#pragma unroll
    for (int t = 0; t < 8; ++t) {
      int c = t * 16 + nn;
      float xl = 0.f, xs = 0.f;
#pragma unroll
      for (int h = 0; h < HH; ++h) {
        xl += (float)outf[((size_t)r * HH + h) * DD + c];
        xs += (float)V[(size_t)r * HD + h * DD + c];
      }
      xl *= 0.25f; xs *= 0.25f;
      out[(size_t)r * DD + c] = alpha * xl + (1.f - alpha) * (xg[t][v] * 0.25f) + beta * xs;
    }
  }
}

// ---- launcher --------------------------------------------------------------
extern "C" void kernel_launch(void* const* d_in, const int* in_sizes, int n_in,
                              void* d_out, int out_size, void* d_ws, size_t ws_size,
                              hipStream_t stream) {
  (void)in_sizes; (void)n_in; (void)out_size; (void)ws_size;
  const float* x    = (const float*)d_in[0];
  const int*   pidx = (const int*)d_in[1];
  const float* Wq   = (const float*)d_in[2];
  const float* bq   = (const float*)d_in[3];
  const float* Wk   = (const float*)d_in[4];
  const float* bk   = (const float*)d_in[5];
  const float* Wv   = (const float*)d_in[6];
  const float* bv   = (const float*)d_in[7];
  const float* al   = (const float*)d_in[8];
  const float* be   = (const float*)d_in[9];
  float* out = (float*)d_out;

  char* ws = (char*)d_ws;
  size_t off = 0;
  h16* W16  = (h16*)(ws + off); off += (size_t)3 * HD * INC * sizeof(h16);
  h16* Qb   = (h16*)(ws + off); off += (size_t)NROWS * HD * sizeof(h16);
  h16* Kb   = (h16*)(ws + off); off += (size_t)NROWS * HD * sizeof(h16);
  h16* Vb   = (h16*)(ws + off); off += (size_t)NROWS * HD * sizeof(h16);
  h16* outf = (h16*)(ws + off); off += (size_t)NROWS * HH * DD * sizeof(h16);
  h16* ckb  = (h16*)(ws + off); off += (size_t)KPART * HD * sizeof(h16);
  h16* cvb  = (h16*)(ws + off); off += (size_t)KPART * HD * sizeof(h16);

  k_cvtW<<<dim3((3 * HD * INC) / 256), dim3(256), 0, stream>>>(Wq, Wk, Wv, W16);
  k_qkv<<<dim3(3, NROWS / 16), dim3(256), 0, stream>>>(x, W16, bq, bk, bv, Qb, Kb, Vb);
  k_fine<<<dim3(KPART * HH * 4), dim3(256), 0, stream>>>(Qb, Kb, Vb, pidx, outf);
  k_cent<<<dim3(KPART), dim3(256), 0, stream>>>(Kb, Vb, pidx, ckb, cvb);
  k_coarse<<<dim3(NROWS / 128), dim3(256), 0, stream>>>(Qb, Vb, ckb, cvb, outf, al, be, out);
}
